// LongformerSelfAttention_65171833749692
// MI455X (gfx1250) — matrix operand order repaired and checked by reading, not compiled, not run on hardware
//
#include <hip/hip_runtime.h>

typedef __attribute__((ext_vector_type(16))) _Float16 v16h;
typedef __attribute__((ext_vector_type(8)))  _Float16 h8;
typedef __attribute__((ext_vector_type(8)))  float    v8f;

#define S_LEN 4096
#define E_DIM 1024
#define H_CNT 16
#define D_DIM 64
#define WIN   256
#define NWIN  16
#define BATCH 4
#define ROWS  (BATCH * S_LEN)   // 16384

static __device__ __forceinline__ v16h cat16(h8 lo, h8 hi) {
  v16h r;
#pragma unroll
  for (int i = 0; i < 8; ++i) { r[i] = lo[i]; r[i + 8] = hi[i]; }
  return r;
}

static __device__ __forceinline__ v8f wmma_f16(v16h a, v16h b, v8f c) {
  // D = A(16x32 f16) * B(32x16 f16) + C(16x16 f32)
  return __builtin_amdgcn_wmma_f32_16x16x32_f16(
      /*neg_a=*/false, a, /*neg_b=*/false, b,
      /*c_mod=*/(short)0, c, /*reuse_a=*/false, /*reuse_b=*/false);
}

// ---------------------------------------------------------------------------
// One-time precision conversion kernels (run once, outside the GEMM hot loop)
// ---------------------------------------------------------------------------
__global__ __launch_bounds__(256)
void cvt_f32_f16_kernel(const float* __restrict__ src,
                        _Float16* __restrict__ dst) {
  size_t base = ((size_t)blockIdx.x * 256 + threadIdx.x) * 4;
  float4 f = *(const float4*)(src + base);
  dst[base + 0] = (_Float16)f.x;
  dst[base + 1] = (_Float16)f.y;
  dst[base + 2] = (_Float16)f.z;
  dst[base + 3] = (_Float16)f.w;
}

// W[k][n] f32  ->  WT[n][k] f16   (coalesced reads, scattered 2B writes; 1x cost)
__global__ __launch_bounds__(256)
void tpose_cvt_kernel(const float* __restrict__ W,
                      _Float16* __restrict__ WT) {
  size_t idx = (size_t)blockIdx.x * 256 + threadIdx.x;   // 0 .. E*E-1
  int kk = (int)(idx >> 10);
  int nn = (int)(idx & (E_DIM - 1));
  WT[(size_t)nn * E_DIM + kk] = (_Float16)W[idx];
}

// ---------------------------------------------------------------------------
// Shared GEMM core: C[128x128] += A[128xE] * B^T (B stored [n][k] f16).
// Software-pipelined: next tile global->regs overlaps current tile WMMAs.
// Per thread stages one 32B chunk of A and of B per k-step.
// ---------------------------------------------------------------------------
static __device__ __forceinline__ void gemm_core(
    const _Float16* __restrict__ aSrc,   // x16 + (m0+tRow)*E + tCol
    const _Float16* __restrict__ bSrc,   // WT  + (n0+tRow)*E + tCol
    _Float16 (&As)[128][32], _Float16 (&Bs)[128][32],
    int tRow, int tCol, int wm, int wn, int hf, int l16,
    v8f (&acc)[2][4]) {
  // stage k0 = 0
  {
    const h8* as = (const h8*)aSrc;
    const h8* bs = (const h8*)bSrc;
    h8* ad = (h8*)&As[tRow][tCol];
    h8* bd = (h8*)&Bs[tRow][tCol];
    ad[0] = as[0]; ad[1] = as[1];
    bd[0] = bs[0]; bd[1] = bs[1];
  }
  __syncthreads();

  for (int k0 = 0; k0 < E_DIM; k0 += 32) {
    const bool more = (k0 + 32) < E_DIM;
    h8 na0, na1, nb0, nb1;
    if (more) {                                   // next tile -> registers
      const h8* as = (const h8*)(aSrc + k0 + 32);
      const h8* bs = (const h8*)(bSrc + k0 + 32);
      na0 = as[0]; na1 = as[1];
      nb0 = bs[0]; nb1 = bs[1];
      if ((k0 + 64) < E_DIM) {                    // prefetch two tiles ahead
        __builtin_prefetch(aSrc + k0 + 64, 0, 3);
        __builtin_prefetch(bSrc + k0 + 64, 0, 3);
      }
    }

    v16h a[2], bfr[4];
#pragma unroll
    for (int mt = 0; mt < 2; ++mt) {
      const _Float16* ap = &As[wm + mt * 16 + l16][hf * 8];
      a[mt] = cat16(*(const h8*)ap, *(const h8*)(ap + 16));
    }
#pragma unroll
    for (int nt = 0; nt < 4; ++nt) {
      const _Float16* bp = &Bs[wn + nt * 16 + l16][hf * 8];
      bfr[nt] = cat16(*(const h8*)bp, *(const h8*)(bp + 16));
    }
#pragma unroll
    for (int mt = 0; mt < 2; ++mt)
#pragma unroll
      for (int nt = 0; nt < 4; ++nt)
        acc[mt][nt] = wmma_f16(a[mt], bfr[nt], acc[mt][nt]);

    __syncthreads();
    if (more) {                                   // regs -> LDS for next step
      h8* ad = (h8*)&As[tRow][tCol];
      h8* bd = (h8*)&Bs[tRow][tCol];
      ad[0] = na0; ad[1] = na1;
      bd[0] = nb0; bd[1] = nb1;
    }
    __syncthreads();
  }
}

// ---------------------------------------------------------------------------
// Kernel 1: fused QKV projection.  y = x16 @ WT^T + b -> [B,H,S,D] f16
// ---------------------------------------------------------------------------
__global__ __launch_bounds__(256)
void qkv_proj_kernel(const _Float16* __restrict__ x16,
                     const _Float16* __restrict__ WTq, const float* __restrict__ bq,
                     const _Float16* __restrict__ WTk, const float* __restrict__ bk,
                     const _Float16* __restrict__ WTv, const float* __restrict__ bv,
                     _Float16* __restrict__ qo, _Float16* __restrict__ ko,
                     _Float16* __restrict__ vo) {
  __shared__ _Float16 As[128][32];
  __shared__ _Float16 Bs[128][32];

  const _Float16* WT; const float* bias; _Float16* dst;
  if (blockIdx.z == 0)      { WT = WTq; bias = bq; dst = qo; }
  else if (blockIdx.z == 1) { WT = WTk; bias = bk; dst = ko; }
  else                      { WT = WTv; bias = bv; dst = vo; }

  const int m0   = blockIdx.y * 128;
  const int n0   = blockIdx.x * 128;
  const int tid  = threadIdx.x;
  const int lane = tid & 31;
  const int wave = tid >> 5;
  const int wm   = (wave >> 1) * 32;
  const int wn   = (wave & 1) * 64;
  const int hf   = lane >> 4;
  const int l16  = lane & 15;
  const int tRow = tid >> 1;
  const int tCol = (tid & 1) * 16;

  v8f acc[2][4];
#pragma unroll
  for (int i = 0; i < 2; ++i)
#pragma unroll
    for (int j = 0; j < 4; ++j) acc[i][j] = (v8f){0,0,0,0,0,0,0,0};

  gemm_core(x16 + (size_t)(m0 + tRow) * E_DIM + tCol,
            WT  + (size_t)(n0 + tRow) * E_DIM + tCol,
            As, Bs, tRow, tCol, wm, wn, hf, l16, acc);

  // epilogue: + bias, scatter to [B,H,S,D] f16
#pragma unroll
  for (int mt = 0; mt < 2; ++mt)
#pragma unroll
    for (int nt = 0; nt < 4; ++nt) {
      int C = n0 + wn + nt * 16 + l16;
      float bb = bias[C];
      int hh = C >> 6, d = C & 63;
#pragma unroll
      for (int r = 0; r < 8; ++r) {
        int R = m0 + wm + mt * 16 + hf * 8 + r;
        int bI = R >> 12;
        int ss = R & (S_LEN - 1);
        size_t off = (((size_t)bI * H_CNT + hh) * S_LEN + ss) * D_DIM + d;
        dst[off] = (_Float16)(acc[mt][nt][r] + bb);
      }
    }
}

// ---------------------------------------------------------------------------
// Kernel 2: block-diagonal window attention, flash style.
// One block per (b, h, window); 8 waves, each owns 32 query rows.
// ---------------------------------------------------------------------------
__global__ __launch_bounds__(256)
void window_attn_kernel(const _Float16* __restrict__ q,
                        const _Float16* __restrict__ k,
                        const _Float16* __restrict__ v,
                        _Float16* __restrict__ ao) {
  __shared__ _Float16 VT[D_DIM][WIN];   // V transposed [d][row]   32 KB
  __shared__ _Float16 Pt[8][32][32];    // per-wave prob tiles     16 KB

  const int blk = blockIdx.x;           // (b*16 + h)*16 + w
  const int wdw = blk & 15;
  const int hh  = (blk >> 4) & 15;
  const int bb  = blk >> 8;

  const size_t base = (size_t)blk * (WIN * D_DIM);
  const _Float16* qb = q + base;
  const _Float16* kb = k + base;
  const _Float16* vb = v + base;

  const int tid  = threadIdx.x;
  const int lane = tid & 31;
  const int wave = tid >> 5;
  const int hf   = lane >> 4;
  const int l16  = lane & 15;

  for (int i = 0; i < 64; ++i) {        // stage V transposed
    int idx = tid + i * 256;
    int r = idx >> 6, d = idx & 63;
    VT[d][r] = vb[idx];
  }
  __syncthreads();

  v16h aq[2][2];                        // persistent Q A-fragments
#pragma unroll
  for (int mt = 0; mt < 2; ++mt)
#pragma unroll
    for (int ks = 0; ks < 2; ++ks) {
      const _Float16* p =
          qb + (size_t)(wave * 32 + mt * 16 + l16) * D_DIM + ks * 32 + hf * 8;
      aq[mt][ks] = cat16(*(const h8*)p, *(const h8*)(p + 16));
    }

  float row_m[16], row_l[16];
#pragma unroll
  for (int i = 0; i < 16; ++i) { row_m[i] = -1e30f; row_l[i] = 0.0f; }

  v8f o[2][4];
#pragma unroll
  for (int mt = 0; mt < 2; ++mt)
#pragma unroll
    for (int nt = 0; nt < 4; ++nt) o[mt][nt] = (v8f){0,0,0,0,0,0,0,0};

  for (int j = 0; j < 8; ++j) {         // 8 blocks of 32 key columns
    if (j < 7)
      __builtin_prefetch(kb + (size_t)((j + 1) * 32 + lane) * D_DIM, 0, 3);

    v8f sc[2][2];
#pragma unroll
    for (int mt = 0; mt < 2; ++mt)
#pragma unroll
      for (int nt = 0; nt < 2; ++nt) sc[mt][nt] = (v8f){0,0,0,0,0,0,0,0};

#pragma unroll
    for (int nt = 0; nt < 2; ++nt)      // scores = Q @ K^T
#pragma unroll
      for (int ks = 0; ks < 2; ++ks) {
        const _Float16* kp =
            kb + (size_t)(j * 32 + nt * 16 + l16) * D_DIM + ks * 32 + hf * 8;
        v16h bk = cat16(*(const h8*)kp, *(const h8*)(kp + 16));
#pragma unroll
        for (int mt = 0; mt < 2; ++mt)
          sc[mt][nt] = wmma_f16(aq[mt][ks], bk, sc[mt][nt]);
      }

    const float scale = 0.125f;         // 1/sqrt(64)
#pragma unroll
    for (int mt = 0; mt < 2; ++mt) {
#pragma unroll
      for (int r = 0; r < 8; ++r) {
        float s0 = sc[mt][0][r] * scale;
        float s1 = sc[mt][1][r] * scale;
        float mx = fmaxf(s0, s1);
#pragma unroll
        for (int msk = 1; msk < 16; msk <<= 1)
          mx = fmaxf(mx, __shfl_xor(mx, msk, 32));
        int st = mt * 8 + r;            // row = mt*16 + hf*8 + r
        float newm  = fmaxf(row_m[st], mx);
        float alpha = __expf(row_m[st] - newm);
        float p0 = __expf(s0 - newm);
        float p1 = __expf(s1 - newm);
        float rs = p0 + p1;
#pragma unroll
        for (int msk = 1; msk < 16; msk <<= 1)
          rs += __shfl_xor(rs, msk, 32);
        row_l[st] = row_l[st] * alpha + rs;
        row_m[st] = newm;
#pragma unroll
        for (int nt = 0; nt < 4; ++nt) o[mt][nt][r] *= alpha;
        Pt[wave][mt * 16 + hf * 8 + r][l16]      = (_Float16)p0;
        Pt[wave][mt * 16 + hf * 8 + r][16 + l16] = (_Float16)p1;
      }
    }

#pragma unroll
    for (int mt = 0; mt < 2; ++mt) {    // O += P @ V
      const _Float16* pp = &Pt[wave][mt * 16 + l16][hf * 8];
      v16h apf = cat16(*(const h8*)pp, *(const h8*)(pp + 16));
#pragma unroll
      for (int nt = 0; nt < 4; ++nt) {
        const _Float16* vp = &VT[nt * 16 + l16][j * 32 + hf * 8];
        v16h bvf = cat16(*(const h8*)vp, *(const h8*)(vp + 16));
        o[mt][nt] = wmma_f16(apf, bvf, o[mt][nt]);
      }
    }
  }

#pragma unroll
  for (int mt = 0; mt < 2; ++mt)        // finalize, store [B,S,E] f16
#pragma unroll
    for (int nt = 0; nt < 4; ++nt)
#pragma unroll
      for (int r = 0; r < 8; ++r) {
        int st = mt * 8 + r;
        float val = o[mt][nt][r] / row_l[st];
        int qrow = wave * 32 + mt * 16 + hf * 8 + r;
        int srow = wdw * WIN + qrow;
        int col  = hh * D_DIM + nt * 16 + l16;
        ao[(size_t)(bb * S_LEN + srow) * E_DIM + col] = (_Float16)val;
      }
}

// ---------------------------------------------------------------------------
// Kernel 3: output projection.  out = attn @ WTo^T + bo  (f32 output)
// ---------------------------------------------------------------------------
__global__ __launch_bounds__(256)
void out_proj_kernel(const _Float16* __restrict__ ain,
                     const _Float16* __restrict__ WTo,
                     const float* __restrict__ bo,
                     float* __restrict__ out) {
  __shared__ _Float16 As[128][32];
  __shared__ _Float16 Bs[128][32];

  const int m0   = blockIdx.y * 128;
  const int n0   = blockIdx.x * 128;
  const int tid  = threadIdx.x;
  const int lane = tid & 31;
  const int wave = tid >> 5;
  const int wm   = (wave >> 1) * 32;
  const int wn   = (wave & 1) * 64;
  const int hf   = lane >> 4;
  const int l16  = lane & 15;
  const int tRow = tid >> 1;
  const int tCol = (tid & 1) * 16;

  v8f acc[2][4];
#pragma unroll
  for (int i = 0; i < 2; ++i)
#pragma unroll
    for (int j = 0; j < 4; ++j) acc[i][j] = (v8f){0,0,0,0,0,0,0,0};

  gemm_core(ain + (size_t)(m0 + tRow) * E_DIM + tCol,
            WTo + (size_t)(n0 + tRow) * E_DIM + tCol,
            As, Bs, tRow, tCol, wm, wn, hf, l16, acc);

#pragma unroll
  for (int mt = 0; mt < 2; ++mt)
#pragma unroll
    for (int nt = 0; nt < 4; ++nt) {
      int C = n0 + wn + nt * 16 + l16;
      float bb = bo[C];
#pragma unroll
      for (int r = 0; r < 8; ++r) {
        int R = m0 + wm + mt * 16 + hf * 8 + r;
        out[(size_t)R * E_DIM + C] = acc[mt][nt][r] + bb;
      }
    }
}

// ---------------------------------------------------------------------------
extern "C" void kernel_launch(void* const* d_in, const int* in_sizes, int n_in,
                              void* d_out, int out_size, void* d_ws, size_t ws_size,
                              hipStream_t stream) {
  (void)in_sizes; (void)n_in; (void)out_size; (void)ws_size;
  const float* x  = (const float*)d_in[0];
  const float* Wq = (const float*)d_in[1];
  const float* bq = (const float*)d_in[2];
  const float* Wk = (const float*)d_in[3];
  const float* bk = (const float*)d_in[4];
  const float* Wv = (const float*)d_in[5];
  const float* bv = (const float*)d_in[6];
  const float* Wo = (const float*)d_in[7];
  const float* bo = (const float*)d_in[8];

  const size_t ELT = (size_t)ROWS * E_DIM;   // 16M elements
  const size_t WEL = (size_t)E_DIM * E_DIM;  // 1M elements
  _Float16* qws = (_Float16*)d_ws;           // [B,H,S,D] f16
  _Float16* kws = qws + ELT;
  _Float16* vws = kws + ELT;
  _Float16* aws = vws + ELT;                 // attention out [B,S,E] f16
  _Float16* x16 = aws + ELT;                 // x converted  [B*S,E] f16
  _Float16* wtq = x16 + ELT;                 // W^T f16 [n][k]
  _Float16* wtk = wtq + WEL;
  _Float16* wtv = wtk + WEL;
  _Float16* wto = wtv + WEL;

  // one-time precision conversion / weight transpose
  cvt_f32_f16_kernel<<<dim3((unsigned)(ELT / 1024)), 256, 0, stream>>>(x, x16);
  tpose_cvt_kernel<<<dim3((unsigned)(WEL / 256)), 256, 0, stream>>>(Wq, wtq);
  tpose_cvt_kernel<<<dim3((unsigned)(WEL / 256)), 256, 0, stream>>>(Wk, wtk);
  tpose_cvt_kernel<<<dim3((unsigned)(WEL / 256)), 256, 0, stream>>>(Wv, wtv);
  tpose_cvt_kernel<<<dim3((unsigned)(WEL / 256)), 256, 0, stream>>>(Wo, wto);

  dim3 g1(E_DIM / 128, ROWS / 128, 3);
  qkv_proj_kernel<<<g1, 256, 0, stream>>>(x16, wtq, bq, wtk, bk, wtv, bv,
                                          qws, kws, vws);

  window_attn_kernel<<<dim3(BATCH * H_CNT * NWIN), 256, 0, stream>>>(
      qws, kws, vws, aws);

  dim3 g3(E_DIM / 128, ROWS / 128);
  out_proj_kernel<<<g3, 256, 0, stream>>>(aws, wto, bo, (float*)d_out);
}
